// ShiftedWindowAttention_57329223467170
// MI455X (gfx1250) — matrix operand order repaired and checked by reading, not compiled, not run on hardware
//
#include <hip/hip_runtime.h>
#include <hip/hip_bf16.h>

typedef _Float16 half16 __attribute__((ext_vector_type(16)));
typedef _Float16 half8  __attribute__((ext_vector_type(8)));
typedef float    float8 __attribute__((ext_vector_type(8)));

#define WSZ   7
#define SHIFT 3
#define NH    3
#define E     96
#define HD    32
#define BATCH 8
#define IMGH  224
#define IMGW  224
#define NWD   32                 // windows per spatial dim
#define NWIN  (BATCH*NWD*NWD)    // 8192
#define NTOK  49
#define NPAD  64

__device__ __forceinline__ half16 cat8(half8 lo, half8 hi) {
  return __builtin_shufflevector(lo, hi, 0,1,2,3,4,5,6,7,8,9,10,11,12,13,14,15);
}

// A fragment (16x32 f16): row-major matrix in LDS, `stride` halves per row.
// ISA: lanes 0-15 hold K=0..7 / 16..23; lanes 16-31 hold K=8..15 / 24..31.
__device__ __forceinline__ half16 frag_a(const _Float16* base, int stride, int m0, int k0, int lane) {
  const _Float16* p = base + (m0 + (lane & 15)) * stride + k0 + ((lane >> 4) << 3);
  half8 lo = *(const half8*)p;          // K = kb .. kb+7
  half8 hi = *(const half8*)(p + 16);   // K = kb+16 .. kb+23
  return cat8(lo, hi);
}

// B fragment (32x16 f16) from transposed store baseT[n][k] (k contiguous).
// ISA: lanes 0-15 hold K=0..7 / 16..23; lanes 16-31 hold K=8..15 / 24..31, column N = lane%16.
__device__ __forceinline__ half16 frag_bT(const _Float16* baseT, int stride, int n0, int k0, int lane) {
  const _Float16* p = baseT + (n0 + (lane & 15)) * stride + k0 + ((lane >> 4) << 3);
  half8 lo = *(const half8*)p;
  half8 hi = *(const half8*)(p + 16);
  return cat8(lo, hi);
}

#define WMMA_F32_F16(a, b, c) \
  __builtin_amdgcn_wmma_f32_16x16x32_f16(false, (a), false, (b), (short)0, (c), false, false)

// ---------------------------------------------------------------------------
// Fully fused shifted-window attention: shift + partition + QKV proj +
// QK^T+bias + softmax + attn*V + out-proj + window reverse + reverse shift.
// One workgroup (192 threads = 6 waves) processes 8 windows; QKV never
// leaves LDS (saves ~460 MB of HBM round-trip vs a two-kernel pipeline).
//
// LDS map (161280 B total -> 2 WGs/WGP):
//   [     0) wqL  f16 [288][96]                         55296
//   [ 55296) woL  f16 [96][96]                          18432
//   [ 73728) bqL  f32 [288]                              1152
//   [ 74880) boL  f32 [96]                                384
//   [ 75264) qkvL f16 [6][64][32] Q,K + [3][32][64] Vt  36864
//   [112128) region1: xL [64][96] -> scH [3][64][64] -> hoL [64][96]  24576
//   [136704) region2: atH [3][64][64] -> foutL f32 [64][96]           24576
// ---------------------------------------------------------------------------
__global__ void __launch_bounds__(192) swin_fused_kernel(
    const float* __restrict__ x, const float* __restrict__ w_qkv,
    const float* __restrict__ b_qkv, const float* __restrict__ w_out,
    const float* __restrict__ b_out, const float* __restrict__ bias_table,
    const int* __restrict__ rel_index, float* __restrict__ out) {
  extern __shared__ char smem[];
  _Float16* wqL   = (_Float16*)(smem);
  _Float16* woL   = (_Float16*)(smem + 55296);
  float*    bqL   = (float*)(smem + 73728);
  float*    boL   = (float*)(smem + 74880);
  _Float16* qkvL  = (_Float16*)(smem + 75264);
  _Float16* xL    = (_Float16*)(smem + 112128);   // region1
  _Float16* scH   = (_Float16*)(smem + 112128);   // region1
  _Float16* hoL   = (_Float16*)(smem + 112128);   // region1
  _Float16* atH   = (_Float16*)(smem + 136704);   // region2
  float*    foutL = (float*)(smem + 136704);      // region2
  _Float16* vtL   = qkvL + 6 * NPAD * HD;         // Vt [3][32][64]

  const int tid  = threadIdx.x;
  const int lane = tid & 31;
  const int wave = tid >> 5;          // 0..5
  const int ln15 = lane & 15;
  const int hi8  = (lane >> 4) << 3;
  const float scale = 0.17677669529663687f;       // 1/sqrt(32)

  // persistent weight staging (once per block, 8 windows amortize it)
  for (int i = tid; i < 288 * E; i += 192) wqL[i] = (_Float16)w_qkv[i];
  for (int i = tid; i < E * E; i += 192)   woL[i] = (_Float16)w_out[i];
  for (int i = tid; i < 288; i += 192)     bqL[i] = b_qkv[i];
  if (tid < E) boL[tid] = b_out[tid];
  if (tid < 38) __builtin_prefetch(rel_index + tid * 64, 0, 2);
  if (tid == 38) __builtin_prefetch(bias_table, 0, 2);

  for (int wi = 0; wi < 8; ++wi) {
    const int win = blockIdx.x * 8 + wi;
    const int b   = win >> 10;
    const int rem = win & 1023;
    const int wh  = rem >> 5;
    const int wwi = rem & 31;

    __syncthreads();   // weights ready / previous window fully consumed

    // stage shifted window tokens into region1 (zero-pad rows 49..63)
    for (int i = tid; i < NPAD * E; i += 192) {
      int tok = i / E, c = i % E;
      _Float16 v = (_Float16)0.f;
      if (tok < NTOK) {
        int r  = tok / WSZ, cc = tok % WSZ;
        int hg = (wh * WSZ + r + SHIFT) % IMGH;   // xs[h,w]=x[(h+s)%H,(w+s)%W]
        int wg = (wwi * WSZ + cc + SHIFT) % IMGW;
        v = (_Float16)x[(((size_t)b * IMGH + hg) * IMGW + wg) * E + c];
      }
      xL[i] = v;
    }
    if (wi < 7 && tid < NTOK) {      // prefetch next window (global_prefetch_b8)
      int win2 = win + 1;
      int b2 = win2 >> 10, rem2 = win2 & 1023;
      int wh2 = rem2 >> 5, ww2 = rem2 & 31;
      int r = tid / WSZ, cc = tid % WSZ;
      int hg = (wh2 * WSZ + r + SHIFT) % IMGH;
      int wg = (ww2 * WSZ + cc + SHIFT) % IMGW;
      __builtin_prefetch(&x[(((size_t)b2 * IMGH + hg) * IMGW + wg) * E], 0, 3);
    }
    __syncthreads();

    // ---- QKV projection straight into LDS (wave owns nt = wave*3..+2) ----
    for (int q = 0; q < 3; ++q) {
      int n0 = (wave * 3 + q) * 16;
      int ch = n0 + ln15;
      float bq = bqL[ch];
      float8 acc[4];
      #pragma unroll
      for (int mt = 0; mt < 4; ++mt)
        #pragma unroll
        for (int r = 0; r < 8; ++r) acc[mt][r] = bq;
      #pragma unroll
      for (int ks = 0; ks < 3; ++ks) {
        half16 bf = frag_bT(wqL, E, n0, ks * 32, lane);   // B(k,n)=w_qkv[n][k]
        #pragma unroll
        for (int mt = 0; mt < 4; ++mt) {
          half16 af = frag_a(xL, E, mt * 16, ks * 32, lane);
          acc[mt] = WMMA_F32_F16(af, bf, acc[mt]);        // 4 independent chains
        }
      }
      int part = ch / E, rm = ch % E, head = rm >> 5, d = rm & 31;
      if (part < 2) {                 // Q (pre-scaled) / K: row-major [tok][d]
        float mul = (part == 0) ? scale : 1.0f;
        _Float16* dstm = qkvL + (part * 3 + head) * NPAD * HD + d;
        #pragma unroll
        for (int mt = 0; mt < 4; ++mt)
          #pragma unroll
          for (int r = 0; r < 8; ++r)
            dstm[(mt * 16 + hi8 + r) * HD] = (_Float16)(acc[mt][r] * mul);
      } else {                        // V written directly transposed [d][tok]
        _Float16* dstm = vtL + (head * HD + d) * NPAD;
        #pragma unroll
        for (int mt = 0; mt < 4; ++mt)
          #pragma unroll
          for (int r = 0; r < 8; ++r)
            dstm[mt * 16 + hi8 + r] = (_Float16)acc[mt][r];
      }
    }
    __syncthreads();

    // ---- scores = q·k^T + bias; wave -> (head = wave/2, mt-half = wave&1) ----
    {
      const int h   = wave >> 1;
      const int mtb = (wave & 1) * 2;
      const _Float16* Qh = qkvL + h * NPAD * HD;
      const _Float16* Kh = qkvL + (3 + h) * NPAD * HD;
      _Float16* sc = scH + h * NPAD * NPAD;
      for (int mi = 0; mi < 2; ++mi) {
        int m0 = (mtb + mi) * 16;
        half16 a = frag_a(Qh, HD, m0, 0, lane);
        #pragma unroll
        for (int nt = 0; nt < 4; ++nt) {
          int n0 = nt * 16;
          int n  = n0 + ln15;
          int nc = n < NTOK - 1 ? n : NTOK - 1;    // clamp (pads unused)
          float8 acc;
          #pragma unroll
          for (int r = 0; r < 8; ++r) {
            int m = m0 + hi8 + r;
            int mc = m < NTOK - 1 ? m : NTOK - 1;
            acc[r] = bias_table[rel_index[mc * NTOK + nc] * NH + h];
          }
          half16 bf = frag_bT(Kh, HD, n0, 0, lane);   // B(k,n)=K[n][k]
          acc = WMMA_F32_F16(a, bf, acc);
          #pragma unroll
          for (int r = 0; r < 8; ++r)
            sc[(m0 + hi8 + r) * NPAD + n] = (_Float16)acc[r];
        }
      }
    }
    __syncthreads();

    // ---- softmax: exactly 192 rows over 192 threads ----
    {
      int h = tid >> 6, m = tid & 63;
      const _Float16* srow = scH + (h * NPAD + m) * NPAD;
      _Float16*       arow = atH + (h * NPAD + m) * NPAD;
      if (m < NTOK) {
        float mx = -3.0e4f;
        for (int n = 0; n < NTOK; ++n) mx = fmaxf(mx, (float)srow[n]);
        float sum = 0.f;
        for (int n = 0; n < NTOK; ++n) {
          float e = __expf((float)srow[n] - mx);
          sum += e;
          arow[n] = (_Float16)e;
        }
        float inv = 1.0f / sum;
        for (int n = 0; n < NTOK; ++n) arow[n] = (_Float16)((float)arow[n] * inv);
        for (int n = NTOK; n < NPAD; ++n) arow[n] = (_Float16)0.f;
      } else {
        for (int n = 0; n < NPAD; ++n) arow[n] = (_Float16)0.f;
      }
    }
    __syncthreads();

    // ---- head_out = attn @ V; same (head, mt-half) split ----
    {
      const int h   = wave >> 1;
      const int mtb = (wave & 1) * 2;
      const _Float16* VhT = vtL + h * HD * NPAD;   // [d][tok]
      const _Float16* A   = atH + h * NPAD * NPAD;
      for (int mi = 0; mi < 2; ++mi) {
        int m0 = (mtb + mi) * 16;
        half16 a0 = frag_a(A, NPAD, m0, 0, lane);
        half16 a1 = frag_a(A, NPAD, m0, 32, lane);
        #pragma unroll
        for (int dt = 0; dt < 2; ++dt) {
          int d0 = dt * 16;
          float8 acc = {};
          acc = WMMA_F32_F16(a0, frag_bT(VhT, NPAD, d0, 0, lane), acc);
          acc = WMMA_F32_F16(a1, frag_bT(VhT, NPAD, d0, 32, lane), acc);
          #pragma unroll
          for (int r = 0; r < 8; ++r)
            hoL[(m0 + hi8 + r) * E + h * HD + d0 + ln15] = (_Float16)acc[r];
        }
      }
    }
    __syncthreads();

    // ---- out-proj: wave owns nt = wave; shared-B, 4-way acc ILP ----
    {
      int n0 = wave * 16;
      int ch = n0 + ln15;
      float bv = boL[ch];
      float8 acc[4];
      #pragma unroll
      for (int mt = 0; mt < 4; ++mt)
        #pragma unroll
        for (int r = 0; r < 8; ++r) acc[mt][r] = bv;
      #pragma unroll
      for (int ks = 0; ks < 3; ++ks) {
        half16 bf = frag_bT(woL, E, n0, ks * 32, lane);   // B(k,n)=w_out[n][k]
        #pragma unroll
        for (int mt = 0; mt < 4; ++mt) {
          half16 af = frag_a(hoL, E, mt * 16, ks * 32, lane);
          acc[mt] = WMMA_F32_F16(af, bf, acc[mt]);
        }
      }
      #pragma unroll
      for (int mt = 0; mt < 4; ++mt)
        #pragma unroll
        for (int r = 0; r < 8; ++r)
          foutL[(mt * 16 + hi8 + r) * E + ch] = acc[mt][r];
    }
    __syncthreads();

    // ---- contiguous b128 NHWC store with window-reverse + reverse shift ----
    for (int i = tid; i < NTOK * 24; i += 192) {    // 49 rows x 24 uint4
      int tok = i / 24, qq = i % 24;
      int ii = tok / WSZ, jj = tok % WSZ;
      int hg = (wh * WSZ + ii + SHIFT) % IMGH;
      int wg = (wwi * WSZ + jj + SHIFT) % IMGW;
      ((uint4*)(out + (((size_t)b * IMGH + hg) * IMGW + wg) * E))[qq] =
          ((const uint4*)(foutL + tok * E))[qq];
    }
  }
}

extern "C" void kernel_launch(void* const* d_in, const int* in_sizes, int n_in,
                              void* d_out, int out_size, void* d_ws, size_t ws_size,
                              hipStream_t stream) {
  const float* x          = (const float*)d_in[0];
  const float* w_qkv      = (const float*)d_in[1];
  const float* b_qkv      = (const float*)d_in[2];
  const float* w_out      = (const float*)d_in[3];
  const float* b_out      = (const float*)d_in[4];
  const float* bias_table = (const float*)d_in[5];
  const int*   rel_index  = (const int*)d_in[6];
  float* out = (float*)d_out;
  (void)d_ws; (void)ws_size;   // fully fused: no global scratch needed

  const size_t smemF = 161280;   // 157.5 KB -> 2 WGs/WGP (12 waves)

  (void)hipFuncSetAttribute(reinterpret_cast<const void*>(swin_fused_kernel),
                            hipFuncAttributeMaxDynamicSharedMemorySize, (int)smemF);

  swin_fused_kernel<<<NWIN / 8, 192, smemF, stream>>>(
      x, w_qkv, b_qkv, w_out, b_out, bias_table, rel_index, out);
}
